// VJEPA21Predictor_45552423141507
// MI455X (gfx1250) — compile-verified, matrix operand order checked by reading, not run on hardware
//
#include <hip/hip_runtime.h>

// ---------------------------------------------------------------------------
// V-JEPA2 encoder layer for MI455X (gfx1250, wave32, WMMA bf16 path).
// GEMMs + attention on v_wmma_f32_16x16x32_bf16; attention softmax stats are
// computed in TRANSPOSED score layout (S' = K*Q^T) so reductions are in-lane
// VALU over vector components + one xor-16 shuffle, not 4-level butterflies.
// GEMM staging uses CDNA5 GLOBAL_LOAD_ASYNC_TO_LDS (ASYNCcnt) when available.
// ---------------------------------------------------------------------------

typedef __attribute__((ext_vector_type(16))) __bf16 v16bf;
typedef __attribute__((ext_vector_type(8)))  float  v8f;
typedef int v4i __attribute__((vector_size(16)));

constexpr int BB   = 4;
constexpr int NTOK = 1568;           // 8*14*14 tokens
constexpr int DM   = 1024;
constexpr int HH   = 16;
constexpr int DH   = 64;
constexpr int DFF  = 4096;
constexpr int MROW = BB * NTOK;      // 6272 = 49 * 128
constexpr float ATT_SCALE = 0.125f;  // 64^-0.5

constexpr int EPI_QK    = 1;   // scatter to [b,h,n,dh] bf16
constexpr int EPI_VT    = 2;   // scatter to [b,h,dh,n] bf16 (transposed V)
constexpr int EPI_RESID = 3;   // f32 out = resid + acc + bias
constexpr int EPI_GELU  = 4;   // bf16 out = gelu(acc + bias)

#if __has_builtin(__builtin_amdgcn_global_load_async_to_lds_b128) && \
    __has_builtin(__builtin_amdgcn_s_wait_asynccnt)
#define USE_ASYNC_LDS 1
typedef __attribute__((address_space(1))) v4i GA4;   // global int4
typedef __attribute__((address_space(3))) v4i LA4;   // LDS int4
#endif

// ---- bf16 <-> f32 (round-to-nearest-even) ---------------------------------
__device__ __forceinline__ __bf16 f2bf(float f) {
  unsigned u = __float_as_uint(f);
  unsigned r = (u + 0x7FFFu + ((u >> 16) & 1u)) >> 16;
  unsigned short s = (unsigned short)r;
  __bf16 b; __builtin_memcpy(&b, &s, 2); return b;
}
__device__ __forceinline__ float bf2f(__bf16 b) {
  unsigned short s; __builtin_memcpy(&s, &b, 2);
  return __uint_as_float(((unsigned)s) << 16);
}

// ---- WMMA fragment loader --------------------------------------------------
// 16-bit A fragment (16x32) / B fragment (32x16 col-major): lane l holds
// row (l&15); with grp=(l>>4): elems 0..7 = K [8g..8g+7], elems 8..15 =
// K [16+8g..16+8g+7].  Caller passes (row_ptr + 8*grp); chunk1 is +16.
__device__ __forceinline__ v16bf frag_ld(const __bf16* p) {
  v16bf f;
  __builtin_memcpy(&f, p, 16);
  __builtin_memcpy(reinterpret_cast<char*>(&f) + 16, p + 16, 16);
  return f;
}

__device__ __forceinline__ v8f bwmma(v16bf a, v16bf b, v8f c) {
  return __builtin_amdgcn_wmma_f32_16x16x32_bf16(
      /*neg_a=*/false, a, /*neg_b=*/false, b,
      /*c_mod=*/(short)0, c, /*reuse_a=*/false, /*reuse_b=*/false);
}

// ---------------------------------------------------------------------------
// Weight convert + transpose: f32 W[K][N] -> bf16 Wt[N][K]
// ---------------------------------------------------------------------------
__global__ __launch_bounds__(256)
void convT_kernel(const float* __restrict__ w, __bf16* __restrict__ wt,
                  int K, int Nn) {
  size_t i = (size_t)blockIdx.x * 256 + threadIdx.x;
  if (i >= (size_t)K * Nn) return;
  int k = (int)(i / Nn), n = (int)(i % Nn);
  wt[(size_t)n * K + k] = f2bf(w[i]);
}

// ---------------------------------------------------------------------------
// LayerNorm (row of 1024) -> bf16
// ---------------------------------------------------------------------------
__global__ __launch_bounds__(256)
void ln_kernel(const float* __restrict__ x, const float* __restrict__ g,
               const float* __restrict__ b, __bf16* __restrict__ out) {
  __shared__ float red[2][8];
  const int row = blockIdx.x;
  const int tid = threadIdx.x;
  const float4 v = reinterpret_cast<const float4*>(x + (size_t)row * DM)[tid];
  float s  = v.x + v.y + v.z + v.w;
  float ss = v.x * v.x + v.y * v.y + v.z * v.z + v.w * v.w;
#pragma unroll
  for (int d = 16; d >= 1; d >>= 1) { s += __shfl_xor(s, d); ss += __shfl_xor(ss, d); }
  if ((tid & 31) == 0) { red[0][tid >> 5] = s; red[1][tid >> 5] = ss; }
  __syncthreads();
  float S = 0.f, SS = 0.f;
#pragma unroll
  for (int w = 0; w < 8; ++w) { S += red[0][w]; SS += red[1][w]; }
  const float mean = S * (1.0f / DM);
  const float var  = SS * (1.0f / DM) - mean * mean;
  const float inv  = rsqrtf(var + 1e-6f);
  const float4 gg = reinterpret_cast<const float4*>(g)[tid];
  const float4 bb = reinterpret_cast<const float4*>(b)[tid];
  __bf16* orow = out + (size_t)row * DM;
  orow[tid * 4 + 0] = f2bf((v.x - mean) * inv * gg.x + bb.x);
  orow[tid * 4 + 1] = f2bf((v.y - mean) * inv * gg.y + bb.y);
  orow[tid * 4 + 2] = f2bf((v.z - mean) * inv * gg.z + bb.z);
  orow[tid * 4 + 3] = f2bf((v.w - mean) * inv * gg.w + bb.w);
}

// ---------------------------------------------------------------------------
// Tiled GEMM: C[M,N] f32 = A[M,K]bf16 * Bt[N,K]bf16 (+bias, epilogues)
// Block 256 thr = 8 waves (2x4); block tile 128x128; wave tile 64x32; K-step 32.
// Staging uses CDNA5 async global->LDS when the builtin is available.
// ---------------------------------------------------------------------------
template <int EPI>
__global__ __launch_bounds__(256)
void gemm_kernel(const __bf16* __restrict__ A, const __bf16* __restrict__ Bt,
                 const float* __restrict__ bias, const float* __restrict__ resid,
                 float* __restrict__ outf, __bf16* __restrict__ outb,
                 int K, int Nn) {
  __shared__ __align__(16) __bf16 As[128 * 32];
  __shared__ __align__(16) __bf16 Bs[128 * 32];
  const int tid  = threadIdx.x;
  const int lane = tid & 31;
  const int wave = tid >> 5;
  const int grp  = lane >> 4;
  const int ln   = lane & 15;
  const int wm   = wave >> 2;           // 0..1  (M half)
  const int wn   = wave & 3;            // 0..3  (N quarter)
  const int m0   = blockIdx.y * 128;
  const int n0   = blockIdx.x * 128;
  const int r     = tid >> 1;           // 0..127 staging row
  const int halfk = (tid & 1) * 16;     // which 16-elem half of the K=32 row

  v8f acc[4][2] = {};

  for (int kk = 0; kk < K; kk += 32) {
    const __bf16* ga = A  + (size_t)(m0 + r) * K + kk + halfk;
    const __bf16* gb = Bt + (size_t)(n0 + r) * K + kk + halfk;
    __bf16* la = As + r * 32 + halfk;
    __bf16* lb = Bs + r * 32 + halfk;
#ifdef USE_ASYNC_LDS
    __builtin_amdgcn_global_load_async_to_lds_b128((GA4*)ga,       (LA4*)la,       0, 0);
    __builtin_amdgcn_global_load_async_to_lds_b128((GA4*)(ga + 8), (LA4*)(la + 8), 0, 0);
    __builtin_amdgcn_global_load_async_to_lds_b128((GA4*)gb,       (LA4*)lb,       0, 0);
    __builtin_amdgcn_global_load_async_to_lds_b128((GA4*)(gb + 8), (LA4*)(lb + 8), 0, 0);
    __builtin_amdgcn_s_wait_asynccnt(0);
#else
    *reinterpret_cast<uint4*>(la)     = *reinterpret_cast<const uint4*>(ga);
    *reinterpret_cast<uint4*>(la + 8) = *reinterpret_cast<const uint4*>(ga + 8);
    *reinterpret_cast<uint4*>(lb)     = *reinterpret_cast<const uint4*>(gb);
    *reinterpret_cast<uint4*>(lb + 8) = *reinterpret_cast<const uint4*>(gb + 8);
#endif
    __syncthreads();
    v16bf af[4], bfr[2];
#pragma unroll
    for (int mi = 0; mi < 4; ++mi)
      af[mi] = frag_ld(As + (wm * 64 + mi * 16 + ln) * 32 + 8 * grp);
#pragma unroll
    for (int ni = 0; ni < 2; ++ni)
      bfr[ni] = frag_ld(Bs + (wn * 32 + ni * 16 + ln) * 32 + 8 * grp);
#pragma unroll
    for (int mi = 0; mi < 4; ++mi)
#pragma unroll
      for (int ni = 0; ni < 2; ++ni)
        acc[mi][ni] = bwmma(af[mi], bfr[ni], acc[mi][ni]);
    __syncthreads();
  }

#pragma unroll
  for (int mi = 0; mi < 4; ++mi)
#pragma unroll
    for (int ni = 0; ni < 2; ++ni)
#pragma unroll
      for (int v = 0; v < 8; ++v) {
        const int row = m0 + wm * 64 + mi * 16 + v + 8 * grp;
        const int col = n0 + wn * 32 + ni * 16 + ln;
        float val = acc[mi][ni][v] + bias[col];
        if constexpr (EPI == EPI_RESID) {
          const size_t o = (size_t)row * Nn + col;
          outf[o] = resid[o] + val;
        } else if constexpr (EPI == EPI_GELU) {
          const float gel = 0.5f * val * (1.0f + erff(val * 0.70710678f));
          outb[(size_t)row * Nn + col] = f2bf(gel);
        } else if constexpr (EPI == EPI_QK) {
          const int bI = row / NTOK, nI = row % NTOK;
          const int hI = col >> 6,  dI = col & 63;
          outb[(((size_t)bI * HH + hI) * NTOK + nI) * DH + dI] = f2bf(val);
        } else if constexpr (EPI == EPI_VT) {
          const int bI = row / NTOK, nI = row % NTOK;
          const int hI = col >> 6,  dI = col & 63;
          outb[(((size_t)bI * HH + hI) * DH + dI) * NTOK + nI] = f2bf(val);
        } else {
          outb[(size_t)row * Nn + col] = f2bf(val);
        }
      }
}

// ---------------------------------------------------------------------------
// 3-axis factorized RoPE on [B,H,N,DH] bf16 (first 60 dims; last 4 untouched)
// ---------------------------------------------------------------------------
__global__ __launch_bounds__(64)
void rope_kernel(__bf16* __restrict__ qk) {
  const int idx = blockIdx.x;           // (b*H + h)*NTOK + n
  const int n   = idx % NTOK;
  const int d   = threadIdx.x;
  if (d >= 60) return;
  __bf16* rowp = qk + (size_t)idx * DH;
  const int seg = d / 20, j = d % 20, i = j >> 1;
  const int rem = n % 196;              // tokens per frame = 14*14
  float pos;
  if (seg == 0)      pos = (float)(n / 196);
  else if (seg == 1) pos = (float)(rem / 14);
  else               pos = (float)(rem % 14);
  const float omega = __expf(-(float)i * 0.9210340372f);  // 10000^{-i/10}
  const float fr = pos * omega;
  const float sn = __sinf(fr), cs = __cosf(fr);
  const float xv = bf2f(rowp[d]);
  const float xp = bf2f(rowp[d ^ 1]);                     // pair partner
  const float y  = (d & 1) ? xp : -xp;                    // rotate-half
  rowp[d] = f2bf(xv * cs + y * sn);                       // loads precede stores
}

// ---------------------------------------------------------------------------
// Flash attention, 1 wave per (b, h, 16-query tile).
// Scores computed TRANSPOSED: S' = K_tile(A) x Q_tile(B) -> C layout has
// queries across lanes (col = lane&15) and keys in vector components
// (row = v + 8*grp). Softmax max/sum: 15 in-lane VALU ops + ONE xor-16
// shuffle per stat per 32-key tile. P' is written to LDS (transposing to
// A-layout) for the PV WMMA against V^T rows.
// ---------------------------------------------------------------------------
__global__ __launch_bounds__(32)
void attn_kernel(const __bf16* __restrict__ q, const __bf16* __restrict__ k,
                 const __bf16* __restrict__ vT, __bf16* __restrict__ ctx) {
  __shared__ __align__(16) __bf16 P[16 * 32];
  const int lane = threadIdx.x;
  const int grp = lane >> 4, ln = lane & 15;
  const int bh = blockIdx.y;
  const int bI = bh / HH, hI = bh % HH;
  const int n0 = blockIdx.x * 16;

  const __bf16* qb = q  + (size_t)bh * NTOK * DH;
  const __bf16* kb = k  + (size_t)bh * NTOK * DH;
  const __bf16* vb = vT + (size_t)bh * DH * NTOK;

  // q fragments (used as B operand: columns = queries, K = dh)
  const v16bf qf0 = frag_ld(qb + (size_t)(n0 + ln) * DH + 8 * grp);
  const v16bf qf1 = frag_ld(qb + (size_t)(n0 + ln) * DH + 32 + 8 * grp);

  v8f acc[4] = {};
  float runmax = -3.0e38f, runsum = 0.0f;   // stats for query column `ln`

  for (int kt = 0; kt < NTOK / 32; ++kt) {
    const int kbase = kt * 32;
    // k fragments (used as A operand: rows = keys, K = dh)
    const v16bf k0a = frag_ld(kb + (size_t)(kbase + ln) * DH + 8 * grp);
    const v16bf k0b = frag_ld(kb + (size_t)(kbase + ln) * DH + 32 + 8 * grp);
    const v16bf k1a = frag_ld(kb + (size_t)(kbase + 16 + ln) * DH + 8 * grp);
    const v16bf k1b = frag_ld(kb + (size_t)(kbase + 16 + ln) * DH + 32 + 8 * grp);
    v8f s0 = {}, s1 = {};
    s0 = bwmma(k0a, qf0, s0); s0 = bwmma(k0b, qf1, s0);   // keys 0..15 (rows)
    s1 = bwmma(k1a, qf0, s1); s1 = bwmma(k1b, qf1, s1);   // keys 16..31

    // --- in-lane tile max over this lane's 16 keys -------------------------
    float tm = -3.0e38f;
#pragma unroll
    for (int i = 0; i < 8; ++i) tm = fmaxf(tm, fmaxf(s0[i], s1[i]));
    tm *= ATT_SCALE;
    tm = fmaxf(tm, __shfl_xor(tm, 16));                   // merge key groups
    const float nm = fmaxf(runmax, tm);
    const float sc = __expf(runmax - nm);
    runmax = nm;

    // --- exponentiate, in-lane sum, write P' (transposed) to LDS -----------
    float ts = 0.0f;
    float p0[8], p1[8];
#pragma unroll
    for (int i = 0; i < 8; ++i) {
      p0[i] = __expf(s0[i] * ATT_SCALE - nm);
      p1[i] = __expf(s1[i] * ATT_SCALE - nm);
      ts += p0[i] + p1[i];
    }
    ts += __shfl_xor(ts, 16);                             // full 32-key sum
    runsum = runsum * sc + ts;
#pragma unroll
    for (int i = 0; i < 8; ++i) {
      P[ln * 32 + (i + 8 * grp)]      = f2bf(p0[i]);      // P[query][key]
      P[ln * 32 + 16 + (i + 8 * grp)] = f2bf(p1[i]);
    }

    // --- rescale ctx accumulator (scale for row v+8*grp lives in lane v+8g)
    float rs[8];
#pragma unroll
    for (int v = 0; v < 8; ++v) rs[v] = __shfl(sc, v + 8 * grp);
#pragma unroll
    for (int t = 0; t < 4; ++t)
#pragma unroll
      for (int v = 0; v < 8; ++v) acc[t][v] *= rs[v];

    __syncthreads();
    const v16bf pf = frag_ld(P + ln * 32 + 8 * grp);      // A-layout 16x32
#pragma unroll
    for (int t = 0; t < 4; ++t) {
      const v16bf vf = frag_ld(vb + (size_t)(16 * t + ln) * NTOK + kbase + 8 * grp);
      acc[t] = bwmma(pf, vf, acc[t]);
    }
    __syncthreads();
  }

  const float inv = 1.0f / runsum;                        // per query col `ln`
  float rv[8];
#pragma unroll
  for (int v = 0; v < 8; ++v) rv[v] = __shfl(inv, v + 8 * grp);
#pragma unroll
  for (int t = 0; t < 4; ++t)
#pragma unroll
    for (int v = 0; v < 8; ++v) {
      const int row = n0 + v + 8 * grp;
      const int dI  = 16 * t + ln;
      ctx[((size_t)bI * NTOK + row) * DM + hI * DH + dI] = f2bf(acc[t][v] * rv[v]);
    }
}

// ---------------------------------------------------------------------------
extern "C" void kernel_launch(void* const* d_in, const int* in_sizes, int n_in,
                              void* d_out, int out_size, void* d_ws, size_t ws_size,
                              hipStream_t stream) {
  (void)in_sizes; (void)n_in; (void)out_size; (void)ws_size;
  const float* x     = (const float*)d_in[0];
  const float* ln1_g = (const float*)d_in[1];
  const float* ln1_b = (const float*)d_in[2];
  const float* wq = (const float*)d_in[3];  const float* bq = (const float*)d_in[4];
  const float* wk = (const float*)d_in[5];  const float* bk = (const float*)d_in[6];
  const float* wv = (const float*)d_in[7];  const float* bv = (const float*)d_in[8];
  const float* wo = (const float*)d_in[9];  const float* bo = (const float*)d_in[10];
  const float* ln2_g = (const float*)d_in[11];
  const float* ln2_b = (const float*)d_in[12];
  const float* w1 = (const float*)d_in[13]; const float* b1 = (const float*)d_in[14];
  const float* w2 = (const float*)d_in[15]; const float* b2 = (const float*)d_in[16];
  float* out = (float*)d_out;

  // Workspace carve-out (~178 MB), 256B aligned chunks.
  char* base = (char*)d_ws;
  size_t off = 0;
  auto alloc = [&](size_t bytes) -> char* {
    char* p = base + off;
    off += (bytes + 255) & ~(size_t)255;
    return p;
  };
  __bf16* wqT = (__bf16*)alloc((size_t)DM * DM * 2);
  __bf16* wkT = (__bf16*)alloc((size_t)DM * DM * 2);
  __bf16* wvT = (__bf16*)alloc((size_t)DM * DM * 2);
  __bf16* woT = (__bf16*)alloc((size_t)DM * DM * 2);
  __bf16* w1T = (__bf16*)alloc((size_t)DFF * DM * 2);
  __bf16* w2T = (__bf16*)alloc((size_t)DM * DFF * 2);
  __bf16* h1   = (__bf16*)alloc((size_t)MROW * DM * 2);
  __bf16* qbuf = (__bf16*)alloc((size_t)MROW * DM * 2);
  __bf16* kbuf = (__bf16*)alloc((size_t)MROW * DM * 2);
  __bf16* vTb  = (__bf16*)alloc((size_t)MROW * DM * 2);
  __bf16* ctx  = (__bf16*)alloc((size_t)MROW * DM * 2);
  float*  x1   = (float*) alloc((size_t)MROW * DM * 4);
  __bf16* h2   = (__bf16*)alloc((size_t)MROW * DM * 2);
  __bf16* a1   = (__bf16*)alloc((size_t)MROW * DFF * 2);

  // 1) weights -> bf16, K-major transposed
  convT_kernel<<<(DM * DM) / 256, 256, 0, stream>>>(wq, wqT, DM, DM);
  convT_kernel<<<(DM * DM) / 256, 256, 0, stream>>>(wk, wkT, DM, DM);
  convT_kernel<<<(DM * DM) / 256, 256, 0, stream>>>(wv, wvT, DM, DM);
  convT_kernel<<<(DM * DM) / 256, 256, 0, stream>>>(wo, woT, DM, DM);
  convT_kernel<<<(DM * DFF) / 256, 256, 0, stream>>>(w1, w1T, DM, DFF);
  convT_kernel<<<(DFF * DM) / 256, 256, 0, stream>>>(w2, w2T, DFF, DM);

  // 2) LN1
  ln_kernel<<<MROW, 256, 0, stream>>>(x, ln1_g, ln1_b, h1);

  // 3) QKV projections (scatter to head layout; V transposed for PV WMMA)
  dim3 g1024(DM / 128, MROW / 128);
  gemm_kernel<EPI_QK><<<g1024, 256, 0, stream>>>(h1, wqT, bq, nullptr, nullptr, qbuf, DM, DM);
  gemm_kernel<EPI_QK><<<g1024, 256, 0, stream>>>(h1, wkT, bk, nullptr, nullptr, kbuf, DM, DM);
  gemm_kernel<EPI_VT><<<g1024, 256, 0, stream>>>(h1, wvT, bv, nullptr, nullptr, vTb, DM, DM);

  // 4) RoPE on q and k
  rope_kernel<<<BB * HH * NTOK, 64, 0, stream>>>(qbuf);
  rope_kernel<<<BB * HH * NTOK, 64, 0, stream>>>(kbuf);

  // 5) attention
  attn_kernel<<<dim3(NTOK / 16, BB * HH), 32, 0, stream>>>(qbuf, kbuf, vTb, ctx);

  // 6) out-proj + residual
  gemm_kernel<EPI_RESID><<<g1024, 256, 0, stream>>>(ctx, woT, bo, x, x1, nullptr, DM, DM);

  // 7) LN2
  ln_kernel<<<MROW, 256, 0, stream>>>(x1, ln2_g, ln2_b, h2);

  // 8) MLP
  gemm_kernel<EPI_GELU><<<dim3(DFF / 128, MROW / 128), 256, 0, stream>>>(
      h2, w1T, b1, nullptr, nullptr, a1, DM, DFF);
  gemm_kernel<EPI_RESID><<<g1024, 256, 0, stream>>>(a1, w2T, b2, x1, out, nullptr, DFF, DM);
}
